// MultiGAT_66932770341054
// MI455X (gfx1250) — compile-verified, matrix operand order checked by reading, not run on hardware
//
#include <hip/hip_runtime.h>
#include <math.h>

// Problem constants (from reference)
#define G_  2
#define N_  20000
#define E_  160000
#define F_  128
#define C_  64
#define H_  8
#define HC_ (H_*C_)     // 512
#define EN_ (E_+N_)     // 180000 edges incl. self loops
#define GN_ (G_*N_)     // 40000
#define NEG_SLOPE 0.2f

typedef __attribute__((ext_vector_type(16))) __bf16 v16bf;
typedef __attribute__((ext_vector_type(8)))  float  v8f;

// ---------------------------------------------------------------- utilities
__device__ __forceinline__ void atomicMaxF(float* addr, float val) {
    unsigned int* ua = reinterpret_cast<unsigned int*>(addr);
    unsigned int cur = __float_as_uint(*addr);
    while (__uint_as_float(cur) < val) {
        unsigned int prev = atomicCAS(ua, cur, __float_as_uint(val));
        if (prev == cur) break;
        cur = prev;
    }
}

__global__ void fill_kernel(float* __restrict__ p, float v, int n) {
    int i = blockIdx.x * blockDim.x + threadIdx.x;
    int stride = gridDim.x * blockDim.x;
    for (; i < n; i += stride) p[i] = v;
}

// ------------------------------------------------- self-loop edge weights
__global__ void deg_kernel(const int* __restrict__ ei, const float* __restrict__ ew,
                           float* __restrict__ cnt, float* __restrict__ sumw) {
    int i = blockIdx.x * blockDim.x + threadIdx.x;
    if (i >= G_ * E_) return;
    int g = i / E_, e = i - g * E_;
    int dst = ei[((size_t)g * 2 + 1) * E_ + e];
    atomicAdd(&cnt[g * N_ + dst], 1.0f);
    atomicAdd(&sumw[g * N_ + dst], ew[(size_t)g * E_ + e]);
}

__global__ void loopdiv_kernel(const float* __restrict__ cnt, float* __restrict__ sumw) {
    int i = blockIdx.x * blockDim.x + threadIdx.x;
    if (i >= GN_) return;
    sumw[i] = sumw[i] / fmaxf(cnt[i], 1.0f);
}

// --------------------------------------------------------- bf16 WMMA GEMM
// C[M x Ncol] = A[M x K] (fp32) @ W[K x Ncol] (fp32) + bias, bf16 compute, fp32 acc.
// Block tile 128(M) x 64(N), K step 32, 8 waves: wave w -> rows [16w,16w+16).
__global__ __launch_bounds__(256) void wmma_gemm_bias(
        const float* __restrict__ A, const float* __restrict__ W,
        const float* __restrict__ bias, float* __restrict__ Cout,
        int M, int K, int Ncol) {
    __shared__ alignas(16) __bf16 As[128][40];   // [row][k], 80B row stride (16B aligned)
    __shared__ alignas(16) __bf16 Bs[64][40];    // transposed: [n][k]

    const int bm   = blockIdx.x * 128;
    const int bn   = blockIdx.y * 64;
    const int tid  = threadIdx.x;
    const int wave = tid >> 5;
    const int lane = tid & 31;
    const int half = lane >> 4;   // 0 or 1
    const int mn   = lane & 15;   // M (A) / N (B) index

    v8f acc[4] = {};

    // A-loader coords (uniform row guard, 16 contiguous k's -> 2x ds_store_b128)
    const int ar   = tid >> 1;
    const int akk0 = (tid & 1) * 16;
    // B-loader coords (8 contiguous n's at fixed k -> 2x global float4 loads)
    const int bkk = (tid * 8) >> 6;
    const int bnn = (tid * 8) & 63;

    for (int kb = 0; kb < K; kb += 32) {
        // ---- stage A tile (128 x 32 fp32 -> bf16)
        {
            union { __bf16 h[16]; uint4 q[2]; } tmp;
            int grow = bm + ar;
            if (grow < M) {
                const float4* s4 = reinterpret_cast<const float4*>(
                        A + (size_t)grow * K + kb + akk0);
                #pragma unroll
                for (int j = 0; j < 4; ++j) {
                    float4 v = s4[j];
                    tmp.h[4*j+0] = (__bf16)v.x;
                    tmp.h[4*j+1] = (__bf16)v.y;
                    tmp.h[4*j+2] = (__bf16)v.z;
                    tmp.h[4*j+3] = (__bf16)v.w;
                }
            } else {
                #pragma unroll
                for (int j = 0; j < 16; ++j) tmp.h[j] = (__bf16)0.0f;
            }
            uint4* d = reinterpret_cast<uint4*>(&As[ar][akk0]);
            d[0] = tmp.q[0];
            d[1] = tmp.q[1];
        }
        // ---- stage B tile (32 x 64 fp32, transposed into Bs[n][k])
        {
            const float* wp = W + (size_t)(kb + bkk) * Ncol + bn + bnn;
            float4 w0 = *reinterpret_cast<const float4*>(wp);
            float4 w1 = *reinterpret_cast<const float4*>(wp + 4);
            Bs[bnn+0][bkk] = (__bf16)w0.x;
            Bs[bnn+1][bkk] = (__bf16)w0.y;
            Bs[bnn+2][bkk] = (__bf16)w0.z;
            Bs[bnn+3][bkk] = (__bf16)w0.w;
            Bs[bnn+4][bkk] = (__bf16)w1.x;
            Bs[bnn+5][bkk] = (__bf16)w1.y;
            Bs[bnn+6][bkk] = (__bf16)w1.z;
            Bs[bnn+7][bkk] = (__bf16)w1.w;
        }
        __syncthreads();

        // ---- fragments (explicit b128 LDS reads, k order 8*half+s for s<8, 16+8*half+(s-8) for s>=8)
        union { v16bf v; uint4 q[2]; } af;
        {
            int mrow = wave * 16 + mn;
            af.q[0] = *reinterpret_cast<const uint4*>(&As[mrow][half * 8]);
            af.q[1] = *reinterpret_cast<const uint4*>(&As[mrow][16 + half * 8]);
        }
        union { v16bf v; uint4 q[2]; } bf[4];
        #pragma unroll
        for (int t = 0; t < 4; ++t) {
            int ncol = t * 16 + mn;
            bf[t].q[0] = *reinterpret_cast<const uint4*>(&Bs[ncol][half * 8]);
            bf[t].q[1] = *reinterpret_cast<const uint4*>(&Bs[ncol][16 + half * 8]);
        }
        #pragma unroll
        for (int t = 0; t < 4; ++t) {
            acc[t] = __builtin_amdgcn_wmma_f32_16x16x32_bf16(
                         false, af.v, false, bf[t].v, (short)0, acc[t], false, false);
        }
        __syncthreads();
    }

    // C/D layout: acc[t][j] -> row = wave*16 + j + 8*half, col = bn + t*16 + mn
    #pragma unroll
    for (int t = 0; t < 4; ++t) {
        int col = bn + t * 16 + mn;
        float bv = bias[col];
        #pragma unroll
        for (int j = 0; j < 8; ++j) {
            int row = bm + wave * 16 + j + half * 8;
            if (row < M) Cout[(size_t)row * Ncol + col] = acc[t][j] + bv;
        }
    }
}

// ------------------------------------------------------ GATv2 edge kernels
// One wave32 per edge. HEADS = 8 (layer 1) or 1 (layer 2); C = 64 (2 ch/lane).
template<int HEADS>
__global__ __launch_bounds__(256) void edge_logits_kernel(
        const int* __restrict__ ei, const float* __restrict__ ew,
        const float* __restrict__ loopw,
        const float* __restrict__ xl, const float* __restrict__ xr,
        const float* __restrict__ We, const float* __restrict__ att,
        float* __restrict__ logit, float* __restrict__ mbuf) {
    int wid  = (blockIdx.x * 256 + threadIdx.x) >> 5;
    int lane = threadIdx.x & 31;
    if (wid >= G_ * EN_) return;
    int g = wid / EN_, e = wid - g * EN_;
    int src, dst; float ea;
    if (e < E_) {
        src = ei[((size_t)g * 2) * E_ + e];
        dst = ei[((size_t)g * 2 + 1) * E_ + e];
        ea  = ew[(size_t)g * E_ + e];
    } else {
        src = dst = e - E_;
        ea = loopw[g * N_ + src];
    }
    const int HCn = HEADS * 64;
    const float* xls = xl + ((size_t)(g * N_ + src)) * HCn;
    const float* xrd = xr + ((size_t)(g * N_ + dst)) * HCn;
    for (int h = 0; h < HEADS; ++h) {
        float part = 0.0f;
        #pragma unroll
        for (int q = 0; q < 2; ++q) {
            int c = h * 64 + q * 32 + lane;
            float s = xls[c] + xrd[c] + ea * We[c];
            s = (s > 0.0f) ? s : NEG_SLOPE * s;
            part += s * att[c];
        }
        #pragma unroll
        for (int off = 16; off > 0; off >>= 1) part += __shfl_xor(part, off, 32);
        if (lane == 0) {
            logit[(size_t)wid * HEADS + h] = part;
            atomicMaxF(&mbuf[((size_t)(g * N_ + dst)) * HEADS + h], part);
        }
    }
}

template<int HEADS>
__global__ void edge_exp_kernel(const int* __restrict__ ei,
                                float* __restrict__ logit,
                                const float* __restrict__ mbuf,
                                float* __restrict__ den) {
    int i = blockIdx.x * blockDim.x + threadIdx.x;
    if (i >= G_ * EN_ * HEADS) return;
    int h  = i % HEADS;
    int ge = i / HEADS;
    int g  = ge / EN_, e = ge - g * EN_;
    int dst = (e < E_) ? ei[((size_t)g * 2 + 1) * E_ + e] : (e - E_);
    float ex = __expf(logit[i] - mbuf[((size_t)(g * N_ + dst)) * HEADS + h]);
    logit[i] = ex;   // in-place: logits -> exp
    atomicAdd(&den[((size_t)(g * N_ + dst)) * HEADS + h], ex);
}

template<int HEADS>
__global__ __launch_bounds__(256) void edge_aggr_kernel(
        const int* __restrict__ ei, const float* __restrict__ exbuf,
        const float* __restrict__ den, const float* __restrict__ xl,
        float* __restrict__ outacc) {
    int wid  = (blockIdx.x * 256 + threadIdx.x) >> 5;
    int lane = threadIdx.x & 31;
    if (wid >= G_ * EN_) return;
    int g = wid / EN_, e = wid - g * EN_;
    int src, dst;
    if (e < E_) {
        src = ei[((size_t)g * 2) * E_ + e];
        dst = ei[((size_t)g * 2 + 1) * E_ + e];
    } else {
        src = dst = e - E_;
    }
    const int HCn = HEADS * 64;
    const float* xls = xl + ((size_t)(g * N_ + src)) * HCn;
    float* od = outacc + ((size_t)(g * N_ + dst)) * HCn;
    for (int h = 0; h < HEADS; ++h) {
        float alpha = exbuf[(size_t)wid * HEADS + h] /
                      den[((size_t)(g * N_ + dst)) * HEADS + h];
        #pragma unroll
        for (int q = 0; q < 2; ++q) {
            int c = h * 64 + q * 32 + lane;
            atomicAdd(&od[c], alpha * xls[c]);
        }
    }
}

__global__ void bias_elu_kernel(float* __restrict__ buf, const float* __restrict__ bias,
                                int cols, int n) {
    int i = blockIdx.x * blockDim.x + threadIdx.x;
    int stride = gridDim.x * blockDim.x;
    for (; i < n; i += stride) {
        float v = buf[i] + bias[i % cols];
        buf[i] = (v > 0.0f) ? v : (__expf(v) - 1.0f);
    }
}

// ----------------------------------------------------------- graph readout
__global__ void gmax_kernel(const float* __restrict__ h2, float* __restrict__ gmax) {
    int t = blockIdx.x * blockDim.x + threadIdx.x;
    int c = t & 63;
    int r0 = t >> 6;
    int rstride = (gridDim.x * blockDim.x) >> 6;
    float m = -1e30f;
    for (int r = r0; r < GN_; r += rstride) m = fmaxf(m, h2[(size_t)r * 64 + c]);
    atomicMaxF(&gmax[c], m);
}

// Gv = relu(gmax @ Wg + bg); gcon = bn + Gv @ Wn[64:128]
__global__ void gv_kernel(const float* __restrict__ gmax, const float* __restrict__ Wg,
                          const float* __restrict__ bg, const float* __restrict__ Wn,
                          const float* __restrict__ bn, float* __restrict__ gcon) {
    __shared__ float gvr[64];
    int c = threadIdx.x;
    if (c < 64) {
        float a = bg[c];
        for (int k = 0; k < 64; ++k) a += gmax[k] * Wg[k * 64 + c];
        gvr[c] = fmaxf(a, 0.0f);
    }
    __syncthreads();
    if (c < 64) {
        float a = bn[c];
        for (int k = 0; k < 64; ++k) a += gvr[k] * Wn[(64 + k) * 64 + c];
        gcon[c] = a;
    }
}

// out[i] = bo + relu(h2[i] @ Wn[0:64] + gcon) . Wo
__global__ __launch_bounds__(128) void final_kernel(
        const float* __restrict__ h2, const float* __restrict__ Wn,
        const float* __restrict__ gcon, const float* __restrict__ Wo,
        const float* __restrict__ bo, float* __restrict__ out) {
    __shared__ float WnT[64 * 64];
    __shared__ float WoS[64];
    __shared__ float gcS[64];
    for (int i = threadIdx.x; i < 64 * 64; i += blockDim.x) WnT[i] = Wn[i];
    if (threadIdx.x < 64) {
        WoS[threadIdx.x] = Wo[threadIdx.x];
        gcS[threadIdx.x] = gcon[threadIdx.x];
    }
    __syncthreads();
    int i = blockIdx.x * blockDim.x + threadIdx.x;
    if (i >= GN_) return;
    float hrow[64];
    const float* hp = h2 + (size_t)i * 64;
    #pragma unroll
    for (int k = 0; k < 64; ++k) hrow[k] = hp[k];
    float o = bo[0];
    for (int c = 0; c < 64; ++c) {
        float a = gcS[c];
        #pragma unroll 8
        for (int k = 0; k < 64; ++k) a += hrow[k] * WnT[k * 64 + c];
        o += fmaxf(a, 0.0f) * WoS[c];
    }
    out[i] = o;
}

// ---------------------------------------------------------------- launcher
extern "C" void kernel_launch(void* const* d_in, const int* in_sizes, int n_in,
                              void* d_out, int out_size, void* d_ws, size_t ws_size,
                              hipStream_t stream) {
    (void)in_sizes; (void)n_in; (void)out_size; (void)ws_size;
    const float* x     = (const float*)d_in[0];
    const int*   ei    = (const int*)  d_in[1];
    const float* ew    = (const float*)d_in[2];
    const float* Wl1   = (const float*)d_in[3];
    const float* bl1   = (const float*)d_in[4];
    const float* Wr1   = (const float*)d_in[5];
    const float* br1   = (const float*)d_in[6];
    const float* We1   = (const float*)d_in[7];
    const float* att1  = (const float*)d_in[8];
    const float* bias1 = (const float*)d_in[9];
    const float* Wl2   = (const float*)d_in[10];
    const float* bl2   = (const float*)d_in[11];
    const float* Wr2   = (const float*)d_in[12];
    const float* br2   = (const float*)d_in[13];
    const float* We2   = (const float*)d_in[14];
    const float* att2  = (const float*)d_in[15];
    const float* bias2 = (const float*)d_in[16];
    const float* Wg    = (const float*)d_in[17];
    const float* bg    = (const float*)d_in[18];
    const float* Wn    = (const float*)d_in[19];
    const float* bn    = (const float*)d_in[20];
    const float* Wo    = (const float*)d_in[21];
    const float* bo    = (const float*)d_in[22];
    float* out = (float*)d_out;
    float* ws  = (float*)d_ws;

    size_t off = 0;
    float* CNT   = ws + off; off += GN_;
    float* LOOPW = ws + off; off += GN_;
    float* XL1   = ws + off; off += (size_t)GN_ * HC_;
    float* XR1   = ws + off; off += (size_t)GN_ * HC_;
    float* H1    = ws + off; off += (size_t)GN_ * HC_;
    float* LOG1  = ws + off; off += (size_t)G_ * EN_ * H_;
    float* M1    = ws + off; off += (size_t)GN_ * H_;
    float* DEN1  = ws + off; off += (size_t)GN_ * H_;
    float* XL2   = ws + off; off += (size_t)GN_ * C_;
    float* XR2   = ws + off; off += (size_t)GN_ * C_;
    float* H2    = ws + off; off += (size_t)GN_ * C_;
    float* LOG2  = ws + off; off += (size_t)G_ * EN_;
    float* M2    = ws + off; off += GN_;
    float* DEN2  = ws + off; off += GN_;
    float* GMAX  = ws + off; off += 64;
    float* GCON  = ws + off; off += 64;

    // re-init all accumulators every call (graph-replay safe)
    fill_kernel<<<512,  256, 0, stream>>>(CNT,  0.0f,   2 * GN_);       // CNT + LOOPW
    fill_kernel<<<4096, 256, 0, stream>>>(H1,   0.0f,   GN_ * HC_);
    fill_kernel<<<512,  256, 0, stream>>>(M1,  -1e30f,  GN_ * H_);
    fill_kernel<<<512,  256, 0, stream>>>(DEN1, 0.0f,   GN_ * H_);
    fill_kernel<<<512,  256, 0, stream>>>(H2,   0.0f,   GN_ * C_);
    fill_kernel<<<64,   256, 0, stream>>>(M2,  -1e30f,  GN_);
    fill_kernel<<<64,   256, 0, stream>>>(DEN2, 0.0f,   GN_);
    fill_kernel<<<1,    64,  0, stream>>>(GMAX,-1e30f,  64);

    // self-loop edge weights
    deg_kernel<<<(G_ * E_ + 255) / 256, 256, 0, stream>>>(ei, ew, CNT, LOOPW);
    loopdiv_kernel<<<(GN_ + 255) / 256, 256, 0, stream>>>(CNT, LOOPW);

    // layer 1 GEMMs (bf16 WMMA)
    dim3 g1((GN_ + 127) / 128, HC_ / 64);
    wmma_gemm_bias<<<g1, 256, 0, stream>>>(x, Wl1, bl1, XL1, GN_, F_, HC_);
    wmma_gemm_bias<<<g1, 256, 0, stream>>>(x, Wr1, br1, XR1, GN_, F_, HC_);

    const int nwb = (G_ * EN_ + 7) / 8;   // waves-per-block = 8
    edge_logits_kernel<H_><<<nwb, 256, 0, stream>>>(ei, ew, LOOPW, XL1, XR1, We1, att1, LOG1, M1);
    edge_exp_kernel<H_><<<(G_ * EN_ * H_ + 255) / 256, 256, 0, stream>>>(ei, LOG1, M1, DEN1);
    edge_aggr_kernel<H_><<<nwb, 256, 0, stream>>>(ei, LOG1, DEN1, XL1, H1);
    bias_elu_kernel<<<4096, 256, 0, stream>>>(H1, bias1, HC_, GN_ * HC_);

    // layer 2 GEMMs (bf16 WMMA)
    dim3 g2((GN_ + 127) / 128, 1);
    wmma_gemm_bias<<<g2, 256, 0, stream>>>(H1, Wl2, bl2, XL2, GN_, HC_, C_);
    wmma_gemm_bias<<<g2, 256, 0, stream>>>(H1, Wr2, br2, XR2, GN_, HC_, C_);

    edge_logits_kernel<1><<<nwb, 256, 0, stream>>>(ei, ew, LOOPW, XL2, XR2, We2, att2, LOG2, M2);
    edge_exp_kernel<1><<<(G_ * EN_ + 255) / 256, 256, 0, stream>>>(ei, LOG2, M2, DEN2);
    edge_aggr_kernel<1><<<nwb, 256, 0, stream>>>(ei, LOG2, DEN2, XL2, H2);
    bias_elu_kernel<<<1024, 256, 0, stream>>>(H2, bias2, C_, GN_ * C_);

    // readout
    gmax_kernel<<<64, 256, 0, stream>>>(H2, GMAX);
    gv_kernel<<<1, 64, 0, stream>>>(GMAX, Wg, bg, Wn, bn, GCON);
    final_kernel<<<(GN_ + 127) / 128, 128, 0, stream>>>(H2, Wn, GCON, Wo, bo, out);
}
